// HybridBlock_50208167690539
// MI455X (gfx1250) — matrix operand order repaired and checked by reading, not compiled, not run on hardware
//
#include <hip/hip_runtime.h>
#include <hip/hip_bf16.h>
#include <stdint.h>

typedef __attribute__((ext_vector_type(16))) _Float16 v16h;
typedef __attribute__((ext_vector_type(8)))  float    v8f;

namespace {
constexpr int kB = 2, kL = 2048, kH = 2560;
constexpr int kDin = 8192, kDg = 4096, kNg = 32, kDs = 64, kHd = 128, kKc = 4, kInter = 9216;
constexpr int kM = kB * kL; // 4096 token rows
}

// ----------------------------- f32 -> f16 -----------------------------------
__global__ void cvt_f16_kernel(const float* __restrict__ s, _Float16* __restrict__ d,
                               size_t n) {
  size_t i = (size_t)blockIdx.x * blockDim.x + threadIdx.x;
  size_t st = (size_t)gridDim.x * blockDim.x;
  for (; i < n; i += st) d[i] = (_Float16)s[i];
}

// ------------------------- RMSNorm (f32 in, f16 out) -------------------------
__global__ __launch_bounds__(256) void rmsnorm_f16_kernel(
    const float* __restrict__ x, const float* __restrict__ w,
    _Float16* __restrict__ out, int n) {
  const int row = blockIdx.x, t = threadIdx.x;
  const float* xr = x + (size_t)row * n;
  float ss = 0.f;
  for (int c = t; c < n; c += 256) { float v = xr[c]; ss += v * v; }
  __shared__ float red[256];
  red[t] = ss; __syncthreads();
  for (int s = 128; s > 0; s >>= 1) { if (t < s) red[t] += red[t + s]; __syncthreads(); }
  const float rs = rsqrtf(red[0] / (float)n + 1e-6f);
  _Float16* orow = out + (size_t)row * n;
  for (int c = t; c < n; c += 256) orow[c] = (_Float16)(xr[c] * rs * w[c]);
}

// ------------------------------- WMMA GEMM -----------------------------------
// C[M,N] = A[M,K] * W[N,K]^T   (M = gridDim.y*128, multiple of 128; K % 64 == 0)
// 128x128 block tile, 8 wave32 (4x2), each wave 32x64 = 2x4 wmma tiles.
// Double-buffered LDS: global loads for panel k+1 issue before panel k compute.
template<bool OUTF16, bool ADDRES>
__global__ __launch_bounds__(256) void gemm_wmma_kernel(
    const _Float16* __restrict__ A, const _Float16* __restrict__ W,
    void* __restrict__ Cv, const float* __restrict__ resid, int N, int K) {
  constexpr int LDT = 40;  // padded LDS row stride in halves (80B -> conflict-free)
  __shared__ __align__(16) _Float16 As[2][128 * LDT];
  __shared__ __align__(16) _Float16 Bs[2][128 * LDT];

  const int tid = threadIdx.x;
  const int lane = tid & 31, wv = tid >> 5;
  const int half = lane >> 4, idx = lane & 15;
  const int wm = wv >> 1, wn = wv & 1;
  const int rowBase = blockIdx.y * 128;
  const int colBase = blockIdx.x * 128;

  // Staging coordinates: thread covers rows r0 and r0+64, segment sg (8 halves).
  const int r0 = tid >> 2, sg = tid & 3;
  const size_t aoff0 = (size_t)(rowBase + r0) * K + sg * 8;
  const size_t aoff1 = (size_t)(rowBase + r0 + 64) * K + sg * 8;
  const int n0 = colBase + r0, n1 = n0 + 64;
  const size_t boff0 = (size_t)n0 * K + sg * 8;
  const size_t boff1 = (size_t)n1 * K + sg * 8;

  uint4 ra0, ra1, rb0, rb1;
  auto gload = [&](int kk) {
    ra0 = *(const uint4*)(A + aoff0 + kk);
    ra1 = *(const uint4*)(A + aoff1 + kk);
    uint4 z = {0u, 0u, 0u, 0u};
    rb0 = z; rb1 = z;
    if (n0 < N) rb0 = *(const uint4*)(W + boff0 + kk);
    if (n1 < N) rb1 = *(const uint4*)(W + boff1 + kk);
  };
  auto lstore = [&](int buf) {
    *(uint4*)(&As[buf][r0 * LDT + sg * 8]) = ra0;
    *(uint4*)(&As[buf][(r0 + 64) * LDT + sg * 8]) = ra1;
    *(uint4*)(&Bs[buf][r0 * LDT + sg * 8]) = rb0;
    *(uint4*)(&Bs[buf][(r0 + 64) * LDT + sg * 8]) = rb1;
  };

  const v8f vzero = {0.f, 0.f, 0.f, 0.f, 0.f, 0.f, 0.f, 0.f};
  v8f acc[2][4];
#pragma unroll
  for (int i = 0; i < 2; ++i)
#pragma unroll
    for (int j = 0; j < 4; ++j) acc[i][j] = vzero;

  union Frag { v16h v; uint4 q[2]; };

  auto compute = [&](int buf) {
    // A fragment (ISA 16-bit 16x32): lane half h holds K[8h..8h+7] and K[16+8h..].
    Frag a[2], b[4];
#pragma unroll
    for (int i = 0; i < 2; ++i) {
      const _Float16* p = &As[buf][(wm * 32 + i * 16 + idx) * LDT + half * 8];
      a[i].q[0] = *(const uint4*)p;
      a[i].q[1] = *(const uint4*)(p + 16);
    }
    // B fragment (32x16): lane = column N, half h holds K[8h..8h+7] and K[16+8h..].
#pragma unroll
    for (int j = 0; j < 4; ++j) {
      const _Float16* p = &Bs[buf][(wn * 64 + j * 16 + idx) * LDT + half * 8];
      b[j].q[0] = *(const uint4*)p;
      b[j].q[1] = *(const uint4*)(p + 16);
    }
#pragma unroll
    for (int i = 0; i < 2; ++i)
#pragma unroll
      for (int j = 0; j < 4; ++j)
        acc[i][j] = __builtin_amdgcn_wmma_f32_16x16x32_f16(
            false, a[i].v, false, b[j].v, (short)0, acc[i][j], false, false);
  };

  // Software pipeline: K % 64 == 0, ping-pong buffers, one barrier per K-step.
  gload(0);
  lstore(0);
  __syncthreads();
  for (int kk = 0; kk < K; kk += 64) {
    gload(kk + 32);          // kk+32 < K always (kk <= K-64)
    compute(0);
    lstore(1);
    __syncthreads();
    const bool more = (kk + 64) < K;
    if (more) {
      gload(kk + 64);
      __builtin_prefetch(A + aoff0 + kk + 96, 0, 1);  // next-next panel hint
      if (n0 < N) __builtin_prefetch(W + boff0 + kk + 96, 0, 1);
    }
    compute(1);
    if (more) lstore(0);
    __syncthreads();
  }

  // C/D layout: vgpr r -> M = r + 8*half, N = lane&15.
#pragma unroll
  for (int i = 0; i < 2; ++i)
#pragma unroll
    for (int j = 0; j < 4; ++j)
#pragma unroll
      for (int r = 0; r < 8; ++r) {
        const int row = rowBase + wm * 32 + i * 16 + half * 8 + r;
        const int col = colBase + wn * 64 + j * 16 + idx;
        if (col < N) {
          const size_t o = (size_t)row * N + col;
          float v = acc[i][j][r];
          if (OUTF16) {
            ((_Float16*)Cv)[o] = (_Float16)v;
          } else {
            if (ADDRES) v += resid[o];
            ((float*)Cv)[o] = v;
          }
        }
      }
}

// -------------------- causal depthwise conv1d (k=4) + SiLU -------------------
__global__ __launch_bounds__(256) void conv_silu_kernel(
    const _Float16* __restrict__ xbc, const float* __restrict__ cw,
    float* __restrict__ out) {
  const size_t i = (size_t)blockIdx.x * 256 + threadIdx.x;  // over kM*kDin
  const int c = (int)(i % kDin);
  const size_t ml = i / kDin;
  const int l = (int)(ml % kL);
  const int b = (int)(ml / kL);
  float acc = 0.f;
#pragma unroll
  for (int j = 0; j < kKc; ++j) {
    const int lj = l - (kKc - 1) + j;
    if (lj >= 0)
      acc += cw[c * kKc + j] * (float)xbc[((size_t)b * kL + lj) * kDin + c];
  }
  out[i] = acc / (1.f + expf(-acc));  // SiLU
}

// ------------------------- sequential selective scan -------------------------
// One workgroup per (group, batch); 128 threads = head_dim lanes; each thread
// carries 64 f32 state values; B_t/C_t broadcast via LDS.
__global__ __launch_bounds__(128) void ssm_scan_kernel(
    const float* __restrict__ xbcc, const float* __restrict__ dtin,
    const float* __restrict__ Din, const float* __restrict__ A_log,
    const float* __restrict__ dt_bias, float* __restrict__ y) {
  const int g = blockIdx.x, b = blockIdx.y, t = threadIdx.x;
  const float Ag = -expf(A_log[g]);
  const float bias = dt_bias[g];
  float h[kDs];
#pragma unroll
  for (int s = 0; s < kDs; ++s) h[s] = 0.f;
  __shared__ float Bsh[kDs], Csh[kDs];
  for (int l = 0; l < kL; ++l) {
    const size_t m = (size_t)b * kL + l;
    const float* rowp = xbcc + m * kDin;
    const float xv = rowp[g * kHd + t];
    if (t < kDs) Bsh[t] = rowp[kDg + g * kDs + t];
    else         Csh[t - kDs] = rowp[kDg + kNg * kDs + g * kDs + (t - kDs)];
    const float dtr = dtin[m * kNg + g] + bias;
    const float dt = (dtr > 20.f) ? dtr : log1pf(expf(dtr));  // softplus
    const float dA = expf(dt * Ag);
    const float Dv = Din[m * kNg + g];
    __syncthreads();
    float accy = 0.f;
    const float dx = dt * xv;
#pragma unroll
    for (int s = 0; s < kDs; ++s) {
      const float hn = dA * h[s] + dx * Bsh[s];
      h[s] = hn;
      accy += hn * Csh[s];
    }
    y[m * kDg + g * kHd + t] = accy + Dv * xv;
    __syncthreads();
  }
}

// ------------------- per-head RMSNorm * SiLU(z), f16 out ---------------------
__global__ __launch_bounds__(128) void head_norm_gate_kernel(
    const float* __restrict__ y, const _Float16* __restrict__ z16,
    const float* __restrict__ nw, _Float16* __restrict__ out) {
  const size_t base = (size_t)blockIdx.x * kHd;  // blockIdx.x over M*NG heads
  const int t = threadIdx.x;
  const float v = y[base + t];
  __shared__ float red[kHd];
  red[t] = v * v; __syncthreads();
  for (int s = kHd / 2; s > 0; s >>= 1) { if (t < s) red[t] += red[t + s]; __syncthreads(); }
  const float rs = rsqrtf(red[0] / (float)kHd + 1e-6f);
  const float zv = (float)z16[base + t];
  const float sz = zv / (1.f + expf(-zv));
  out[base + t] = (_Float16)(v * rs * nw[t] * sz);
}

// --------------------------------- SwiGLU ------------------------------------
__global__ __launch_bounds__(256) void swiglu_kernel(
    const _Float16* __restrict__ g16, const _Float16* __restrict__ u16,
    _Float16* __restrict__ act, size_t n) {
  const size_t i = (size_t)blockIdx.x * 256 + threadIdx.x;
  if (i >= n) return;
  const float gv = (float)g16[i];
  const float uv = (float)u16[i];
  act[i] = (_Float16)(gv / (1.f + expf(-gv)) * uv);
}

// ------------------------------- launcher ------------------------------------
extern "C" void kernel_launch(void* const* d_in, const int* in_sizes, int n_in,
                              void* d_out, int out_size, void* d_ws, size_t ws_size,
                              hipStream_t stream) {
  (void)in_sizes; (void)n_in; (void)out_size; (void)ws_size;
  const float* x      = (const float*)d_in[0];
  const float* Wz     = (const float*)d_in[1];
  const float* Wqkv   = (const float*)d_in[2];
  const float* Wb     = (const float*)d_in[3];
  const float* Wa     = (const float*)d_in[4];
  const float* conv_w = (const float*)d_in[5];
  const float* Wout   = (const float*)d_in[6];
  const float* norm_w = (const float*)d_in[7];
  const float* A_log  = (const float*)d_in[8];
  const float* dt_b   = (const float*)d_in[9];
  const float* ln1_w  = (const float*)d_in[10];
  const float* ln2_w  = (const float*)d_in[11];
  const float* Wgate  = (const float*)d_in[12];
  const float* Wup    = (const float*)d_in[13];
  const float* Wdown  = (const float*)d_in[14];
  float* out = (float*)d_out;

  char* ws = (char*)d_ws;
  size_t off = 0;
  auto take = [&](size_t bytes) -> char* {
    char* p = ws + off;
    off = (off + bytes + 255) & ~(size_t)255;
    return p;
  };
  _Float16* wz16   = (_Float16*)take((size_t)kDg * kH * 2);
  _Float16* wqkv16 = (_Float16*)take((size_t)kDin * kH * 2);
  _Float16* wb16   = (_Float16*)take((size_t)kNg * kH * 2);
  _Float16* wa16   = (_Float16*)take((size_t)kNg * kH * 2);
  _Float16* wout16 = (_Float16*)take((size_t)kH * kDg * 2);
  _Float16* wg16   = (_Float16*)take((size_t)kInter * kH * 2);
  _Float16* wu16   = (_Float16*)take((size_t)kInter * kH * 2);
  _Float16* wd16   = (_Float16*)take((size_t)kH * kInter * 2);
  _Float16* xn16   = (_Float16*)take((size_t)kM * kH * 2);
  _Float16* z16    = (_Float16*)take((size_t)kM * kDg * 2);
  _Float16* xbc16  = (_Float16*)take((size_t)kM * kDin * 2);
  float*    dtf    = (float*)take((size_t)kM * kNg * 4);
  float*    Df     = (float*)take((size_t)kM * kNg * 4);
  float*    xbcc   = (float*)take((size_t)kM * kDin * 4);
  float*    yv     = (float*)take((size_t)kM * kDg * 4);
  _Float16* yz16   = (_Float16*)take((size_t)kM * kDg * 2);
  // Reuse (disjoint lifetimes): gate in xbcc (dead after scan), up in y/yz
  // (dead after Wout GEMM), act in z/xbc (dead by MLP stage).
  _Float16* gate16 = (_Float16*)xbcc;
  _Float16* up16   = (_Float16*)yv;
  _Float16* act16  = (_Float16*)z16;

  auto cvt = [&](const float* s, _Float16* d, size_t n) {
    unsigned blocks = (unsigned)((n + 1023) / 1024);
    if (blocks > 65535u) blocks = 65535u;
    cvt_f16_kernel<<<blocks, 256, 0, stream>>>(s, d, n);
  };
  cvt(Wz, wz16, (size_t)kDg * kH);
  cvt(Wqkv, wqkv16, (size_t)kDin * kH);
  cvt(Wb, wb16, (size_t)kNg * kH);
  cvt(Wa, wa16, (size_t)kNg * kH);
  cvt(Wout, wout16, (size_t)kH * kDg);
  cvt(Wgate, wg16, (size_t)kInter * kH);
  cvt(Wup, wu16, (size_t)kInter * kH);
  cvt(Wdown, wd16, (size_t)kH * kInter);

  auto grid2 = [](int N) { return dim3((unsigned)((N + 127) / 128), kM / 128); };

  // (1) xn = rmsnorm(x, ln1_w)
  rmsnorm_f16_kernel<<<kM, 256, 0, stream>>>(x, ln1_w, xn16, kH);
  // (2..5) input projections
  gemm_wmma_kernel<true,  false><<<grid2(kDg),  256, 0, stream>>>(xn16, wz16,   (void*)z16,   nullptr, kDg,  kH);
  gemm_wmma_kernel<true,  false><<<grid2(kDin), 256, 0, stream>>>(xn16, wqkv16, (void*)xbc16, nullptr, kDin, kH);
  gemm_wmma_kernel<false, false><<<grid2(kNg),  256, 0, stream>>>(xn16, wb16,   (void*)dtf,   nullptr, kNg,  kH);
  gemm_wmma_kernel<false, false><<<grid2(kNg),  256, 0, stream>>>(xn16, wa16,   (void*)Df,    nullptr, kNg,  kH);
  // (6) causal depthwise conv + SiLU
  conv_silu_kernel<<<(kM * kDin) / 256, 256, 0, stream>>>(xbc16, conv_w, xbcc);
  // (7) sequential SSM scan
  ssm_scan_kernel<<<dim3(kNg, kB), 128, 0, stream>>>(xbcc, dtf, Df, A_log, dt_b, yv);
  // (8) per-head RMSNorm * SiLU(z)
  head_norm_gate_kernel<<<kM * kNg, kHd, 0, stream>>>(yv, z16, norm_w, yz16);
  // (9) output projection + residual -> d_out holds x2
  gemm_wmma_kernel<false, true><<<grid2(kH), 256, 0, stream>>>(yz16, wout16, (void*)out, x, kH, kDg);
  // (10) second RMSNorm
  rmsnorm_f16_kernel<<<kM, 256, 0, stream>>>(out, ln2_w, xn16, kH);
  // (11,12) MLP gate/up projections
  gemm_wmma_kernel<true, false><<<grid2(kInter), 256, 0, stream>>>(xn16, wg16, (void*)gate16, nullptr, kInter, kH);
  gemm_wmma_kernel<true, false><<<grid2(kInter), 256, 0, stream>>>(xn16, wu16, (void*)up16,   nullptr, kInter, kH);
  // (13) SwiGLU
  {
    const size_t n = (size_t)kM * kInter;
    swiglu_kernel<<<(unsigned)((n + 255) / 256), 256, 0, stream>>>(gate16, up16, act16, n);
  }
  // (14) down projection + residual (reads x2 from d_out, writes final result)
  gemm_wmma_kernel<false, true><<<grid2(kH), 256, 0, stream>>>(act16, wd16, (void*)out, out, kH, kInter);
}
